// Model2_48747878809739
// MI455X (gfx1250) — matrix operand order repaired and checked by reading, not compiled, not run on hardware
//
#include <hip/hip_runtime.h>

typedef __bf16 bf16;
typedef __attribute__((ext_vector_type(16))) __bf16 bf16x16;
typedef __attribute__((ext_vector_type(8)))  __bf16 bf16x8;
typedef __attribute__((ext_vector_type(8)))  float  v8f;

// Problem constants
#define BB 64
#define TT 512
#define II 64
#define HH 1024
#define FH 4096           // 4*H
#define KE 1088           // I + H (encoder fused K)
#define DSTEPS 48
#define OO 64
#define OUTW (DSTEPS*OO)  // 3072

// Persistent kernel config: 128 WGs x 4 waves; wave owns a 16(M) x 32(N) gate tile
#define GRID_WGS 128
#define NTHR 128
#define EW_STRIDE (GRID_WGS * NTHR)   // 16384

__device__ __forceinline__ float sigmoidf(float x) { return 1.0f / (1.0f + __expf(-x)); }

__device__ __forceinline__ bf16x16 cat8(bf16x8 lo, bf16x8 hi) {
  return __builtin_shufflevector(lo, hi, 0,1,2,3,4,5,6,7,8,9,10,11,12,13,14,15);
}
// A fragment (16x32 bf16): lane holds K {kA..kA+7, 16+kA..16+kA+7} of its row
__device__ __forceinline__ bf16x16 load_frag_a(const bf16* __restrict__ p, int kA) {
  return cat8(*(const bf16x8*)(p + kA), *(const bf16x8*)(p + 16 + kA));
}
// B fragment (32x16 bf16): lane holds K {kB..kB+7, 16+kB..16+kB+7} of its column's weight row
__device__ __forceinline__ bf16x16 load_frag_b(const bf16* p, int kB) {
  return cat8(*(const bf16x8*)(p + kB), *(const bf16x8*)(p + 16 + kB));
}
__device__ __forceinline__ v8f wmma(v8f acc, bf16x16 a, bf16x16 b) {
  return __builtin_amdgcn_wmma_f32_16x16x32_bf16(false, a, false, b, (short)0, acc,
                                                 false, false);
}

// Software grid barrier (counter + generation), GRID_WGS workgroups
__device__ __forceinline__ void grid_sync(unsigned* bar) {
  __syncthreads();
  if (threadIdx.x == 0) {
    __threadfence();
    volatile unsigned* vbar = (volatile unsigned*)bar;
    unsigned gen = vbar[1];
    if (atomicAdd(&bar[0], 1u) == (unsigned)(GRID_WGS - 1)) {
      vbar[0] = 0u;
      __threadfence();
      vbar[1] = gen + 1u;
    } else {
      while (vbar[1] == gen) { __builtin_amdgcn_s_sleep(1); }
    }
    __threadfence();
  }
  __syncthreads();
}

// ---------------------------------------------------------------------------
// Prep: x->bf16, Wfc->bf16, xin = y0@Wih_d^T + (bih_d+bhh_d), be = bih_e+bhh_e,
//       zero h (double buffered), c, barrier.
// ---------------------------------------------------------------------------
__global__ void k_prep(const float* __restrict__ x, const float* __restrict__ y0,
                       const float* __restrict__ Wih_d, const float* __restrict__ bih_d,
                       const float* __restrict__ bhh_d, const float* __restrict__ Wfc,
                       const float* __restrict__ bih_e, const float* __restrict__ bhh_e,
                       bf16* __restrict__ xbf, bf16* __restrict__ wfcbf,
                       float* __restrict__ xin, float* __restrict__ be,
                       bf16* __restrict__ h0, bf16* __restrict__ h1,
                       float* __restrict__ cbuf, unsigned* __restrict__ bar) {
  int idx = blockIdx.x * blockDim.x + threadIdx.x;
  if (idx < BB * TT * II) xbf[idx] = (bf16)x[idx];
  if (idx < OO * HH) wfcbf[idx] = (bf16)Wfc[idx];
  if (idx < BB * HH) {
    cbuf[idx] = 0.0f;
    h0[idx] = (bf16)0.0f;
    h1[idx] = (bf16)0.0f;
  }
  if (idx < FH) be[idx] = bih_e[idx] + bhh_e[idx];
  if (idx < BB * FH) {
    int b = idx >> 12, n = idx & (FH - 1);
    float s = bih_d[n] + bhh_d[n];
    const float* yr = y0 + b * II;
    const float* wr = Wih_d + n * II;
#pragma unroll 8
    for (int i = 0; i < II; ++i) s += yr[i] * wr[i];
    xin[idx] = s;
  }
  if (idx < 4) bar[idx] = 0u;
}

// ---------------------------------------------------------------------------
// Persistent LSTM kernel. WG g owns gate columns [32g, 32g+32); wave w (0..3)
// owns batch rows [16w, 16w+16) x both 16-column groups (2 accumulators share
// one A fragment per K-chunk). Weights live in LDS for the whole kernel.
// ---------------------------------------------------------------------------
struct __align__(16) SW {
  bf16 we[32][KE];   // [Wih_e | Whh_e] rows 32g..32g+31  (69,632 B)
  bf16 wd[32][HH];   // Whh_d rows 32g..32g+31            (65,536 B)
};

__global__ __launch_bounds__(NTHR) void k_lstm(
    const float* __restrict__ Wih_e, const float* __restrict__ Whh_e,
    const float* __restrict__ Whh_d, const float* __restrict__ bfc,
    const bf16* __restrict__ xbf, const bf16* __restrict__ wfcbf,
    const float* __restrict__ xin, const float* __restrict__ be,
    bf16* __restrict__ h0, bf16* __restrict__ h1,
    float* __restrict__ cbuf, float* __restrict__ gates,
    unsigned* __restrict__ bar, float* __restrict__ out) {
  __shared__ SW s;
  const int tid = threadIdx.x;
  const int g = blockIdx.x;

  // --- one-time LDS weight stage (fp32 -> bf16) ---
  for (int e = tid; e < 32 * KE; e += NTHR) {
    int r = e / KE, k = e - r * KE;
    int n = g * 32 + r;
    float v = (k < II) ? Wih_e[n * II + k] : Whh_e[n * HH + (k - II)];
    s.we[r][k] = (bf16)v;
  }
  for (int e = tid; e < 32 * HH; e += NTHR) {
    int r = e >> 10, k = e & (HH - 1);
    s.wd[r][k] = (bf16)Whh_d[(g * 32 + r) * HH + k];
  }
  __syncthreads();

  const int lane = tid & 31;
  const int wave = tid >> 5;              // 0..3 = batch tile
  const int gwave = g * 4 + wave;
  const int m0 = wave * 16;               // batch rows
  const int n0g = g * 32;                 // global gate-column base (2 groups of 16)
  const int r = lane & 15;
  const int kA = (lane < 16) ? 0 : 8;     // A-frag K sub-offset
  const int kB = (lane < 16) ? 0 : 8;     // B-frag K sub-offset
  const int roff = (lane < 16) ? 0 : 8;   // C/D row sub-offset
  const int ewbase = g * NTHR + tid;      // 0..16383 elementwise base

  // ---- GEMM for one encoder step: gates = [x_t | h] @ [Wih_e|Whh_e]^T ----
  auto gemm_enc = [&](const bf16* __restrict__ hin, int t) {
    v8f a0 = {}, a1 = {};
    const bf16* w0 = &s.we[r][0];
    const bf16* w1 = &s.we[16 + r][0];
    const bf16* xrow = xbf + ((size_t)(m0 + r) * TT + t) * II;
#pragma unroll
    for (int kc = 0; kc < II; kc += 32) {
      bf16x16 a = load_frag_a(xrow + kc, kA);
      a0 = wmma(a0, a, load_frag_b(w0 + kc, kB));
      a1 = wmma(a1, a, load_frag_b(w1 + kc, kB));
    }
    const bf16* hrow = hin + (size_t)(m0 + r) * HH;
#pragma unroll 2
    for (int kc = 0; kc < HH; kc += 32) {
      bf16x16 a = load_frag_a(hrow + kc, kA);
      a0 = wmma(a0, a, load_frag_b(w0 + II + kc, kB));
      a1 = wmma(a1, a, load_frag_b(w1 + II + kc, kB));
    }
    float* gb = gates + (size_t)(m0 + roff) * FH + n0g + r;
#pragma unroll
    for (int i = 0; i < 8; ++i) {
      gb[(size_t)i * FH] = a0[i];
      gb[(size_t)i * FH + 16] = a1[i];
    }
  };

  // ---- GEMM for one decoder step: gates = h @ Whh_d^T ----
  auto gemm_dec = [&](const bf16* __restrict__ hin) {
    v8f a0 = {}, a1 = {};
    const bf16* w0 = &s.wd[r][0];
    const bf16* w1 = &s.wd[16 + r][0];
    const bf16* hrow = hin + (size_t)(m0 + r) * HH;
#pragma unroll 2
    for (int kc = 0; kc < HH; kc += 32) {
      bf16x16 a = load_frag_a(hrow + kc, kA);
      a0 = wmma(a0, a, load_frag_b(w0 + kc, kB));
      a1 = wmma(a1, a, load_frag_b(w1 + kc, kB));
    }
    float* gb = gates + (size_t)(m0 + roff) * FH + n0g + r;
#pragma unroll
    for (int i = 0; i < 8; ++i) {
      gb[(size_t)i * FH] = a0[i];
      gb[(size_t)i * FH + 16] = a1[i];
    }
  };

  // ---- elementwise LSTM cell: addv strided by bstride per batch row ----
  auto update = [&](const float* __restrict__ addv, int bstride,
                    bf16* __restrict__ hout) {
#pragma unroll
    for (int e = 0; e < 4; ++e) {
      int idx = ewbase + e * EW_STRIDE;
      int b = idx >> 10, j = idx & (HH - 1);
      const float* gr = gates + (size_t)b * FH + j;
      const float* av = addv + (size_t)b * bstride + j;
      float gi = sigmoidf(gr[0]      + av[0]);
      float gf = sigmoidf(gr[HH]     + av[HH]);
      float gg = tanhf   (gr[2 * HH] + av[2 * HH]);
      float go = sigmoidf(gr[3 * HH] + av[3 * HH]);
      float c2 = gf * cbuf[idx] + gi * gg;
      cbuf[idx] = c2;
      hout[idx] = (bf16)(go * tanhf(c2));
    }
  };

  // ---- FC projection for one decoder output (WGs 0..3, 16 waves total) ----
  auto fc = [&](const bf16* __restrict__ hsrc, int st) {
    if (gwave < 16) {
      int fm0 = (gwave & 3) * 16;
      int fn0 = (gwave >> 2) * 16;
      v8f fa = {};
      const bf16* hr2 = hsrc + (size_t)(fm0 + r) * HH;
      const bf16* wr2 = wfcbf + (size_t)(fn0 + r) * HH;
#pragma unroll 4
      for (int kc = 0; kc < HH; kc += 32)
        fa = wmma(fa, load_frag_a(hr2 + kc, kA), load_frag_b(wr2 + kc, kB));
      float bias = bfc[fn0 + r];
      float* ob = out + (size_t)(fm0 + roff) * OUTW + st * OO + fn0 + r;
#pragma unroll
      for (int i = 0; i < 8; ++i) ob[(size_t)i * OUTW] = fa[i] + bias;
    }
  };

  // ================= Encoder: 512 steps (ping-pong h0/h1, static ptrs) ====
  for (int t = 0; t < TT; t += 2) {
    gemm_enc(h0, t);
    grid_sync(bar);
    update(be, 0, h1);
    grid_sync(bar);
    gemm_enc(h1, t + 1);
    grid_sync(bar);
    update(be, 0, h0);
    grid_sync(bar);
  }

  // ================= Decoder: 48 steps =================
  for (int st = 0; st < DSTEPS; st += 2) {
    gemm_dec(h0);
    grid_sync(bar);
    update(xin, FH, h1);
    grid_sync(bar);
    fc(h1, st);          // overlaps next GEMM in non-FC waves
    gemm_dec(h1);
    grid_sync(bar);
    update(xin, FH, h0);
    grid_sync(bar);
    fc(h0, st + 1);
  }
}

// ---------------------------------------------------------------------------
extern "C" void kernel_launch(void* const* d_in, const int* in_sizes, int n_in,
                              void* d_out, int out_size, void* d_ws, size_t ws_size,
                              hipStream_t stream) {
  const float* x     = (const float*)d_in[0];
  const float* y0    = (const float*)d_in[1];
  const float* Wih_e = (const float*)d_in[2];
  const float* Whh_e = (const float*)d_in[3];
  const float* bih_e = (const float*)d_in[4];
  const float* bhh_e = (const float*)d_in[5];
  const float* Wih_d = (const float*)d_in[6];
  const float* Whh_d = (const float*)d_in[7];
  const float* bih_d = (const float*)d_in[8];
  const float* bhh_d = (const float*)d_in[9];
  const float* Wfc   = (const float*)d_in[10];
  const float* bfc   = (const float*)d_in[11];

  char* ws = (char*)d_ws;
  size_t off = 0;
  bf16*     xbf   = (bf16*)(ws + off); off += (size_t)BB * TT * II * 2;   // 4,194,304
  bf16*     wfcbf = (bf16*)(ws + off); off += (size_t)OO * HH * 2;        //   131,072
  float*    xin   = (float*)(ws + off); off += (size_t)BB * FH * 4;       // 1,048,576
  float*    gates = (float*)(ws + off); off += (size_t)BB * FH * 4;       // 1,048,576
  bf16*     h0    = (bf16*)(ws + off); off += (size_t)BB * HH * 2;        //   131,072
  bf16*     h1    = (bf16*)(ws + off); off += (size_t)BB * HH * 2;        //   131,072
  float*    cbuf  = (float*)(ws + off); off += (size_t)BB * HH * 4;       //   262,144
  float*    be    = (float*)(ws + off); off += (size_t)FH * 4;            //    16,384
  unsigned* bar   = (unsigned*)(ws + off); off += 256;

  k_prep<<<(BB * TT * II + 255) / 256, 256, 0, stream>>>(
      x, y0, Wih_d, bih_d, bhh_d, Wfc, bih_e, bhh_e,
      xbf, wfcbf, xin, be, h0, h1, cbuf, bar);

  k_lstm<<<GRID_WGS, NTHR, 0, stream>>>(
      Wih_e, Whh_e, Whh_d, bfc, xbf, wfcbf, xin, be,
      h0, h1, cbuf, gates, bar, (float*)d_out);
}
